// IpaMultiRigidDenoiser_60146722013706
// MI455X (gfx1250) — matrix operand order repaired and checked by reading, not compiled, not run on hardware
//
#include <hip/hip_runtime.h>
#include <hip/hip_bf16.h>
#include <math.h>

typedef _Float16 half_t;
typedef __attribute__((ext_vector_type(16))) _Float16 v16h;
typedef __attribute__((ext_vector_type(8)))  float    v8f;

#define L_TOT  1152
#define NRES_  384
#define CS_    256
#define CZ_    128
#define CH_    16
#define NH     8
#define PQ_    8
#define PV_    12
#define NPROJ  1056          // 128+128+128+192+192+288 (valid columns)
#define NPROJ_LD 1088        // padded to 68 N-tiles for guard-free GEMM
#define CATK   1536          // H*(CH + PV*3 + PV + CZ) = 8*192
#define MT_L   72            // L/16
#define JK_L   36            // L/32

// ---------------- WMMA fragment index maps (CDNA5 ISA 7.12.2, wave32) ----------------
// A matrix 16x32 f16: lane = m + 16*((k>>3)&1); element e = (k&7) + 8*(k>>4)
__device__ __forceinline__ int a_frag_idx(int m, int k) {
  int lane = m + 16 * ((k >> 3) & 1);
  int e = (k & 7) + ((k >> 4) << 3);
  return lane * 16 + e;
}
// B matrix 32x16 f16: lane = n + 16*((k>>3)&1); element e = (k&7) + 8*(k>>4)
__device__ __forceinline__ int b_frag_idx(int k, int n) {
  return (n + 16 * ((k >> 3) & 1)) * 16 + (k & 7) + ((k >> 4) << 3);
}
// C/D 16x16 f32: lane = n + 16*(m>>3); vgpr = m&7

#define WMMA_F16(a, b, c) \
  __builtin_amdgcn_wmma_f32_16x16x32_f16(false, (a), false, (b), (short)0, (c), false, false)

// ---------------- block reduction helper ----------------
__device__ __forceinline__ float block_reduce_sum(float v, float* red, int nwarp) {
  for (int o = 16; o > 0; o >>= 1) v += __shfl_down(v, o, 32);
  int t = threadIdx.x;
  if ((t & 31) == 0) red[t >> 5] = v;
  __syncthreads();
  float s = (t < nwarp) ? red[t] : 0.0f;
  if (t < 32) {
    for (int o = 8; o > 0; o >>= 1) s += __shfl_down(s, o, 32);
  }
  if (t == 0) red[0] = s;
  __syncthreads();
  float r = red[0];
  __syncthreads();
  return r;
}

// ---------------- fragment GEMM: pre-swizzled operands, no LDS, no predication -------
// Af: [Mtiles][Ksteps][512] per batch ; Bf: [Ksteps][Ntiles][512] per batch
// one wave computes a 16 x 64 strip of C (exactly 4 N-tiles; Ntiles % 4 == 0)
__global__ __launch_bounds__(32) void gemm_frag(
    const half_t* __restrict__ Af, const half_t* __restrict__ Bf,
    float* __restrict__ C, const float* __restrict__ bias,
    int Ntiles, int Ksteps, int ldc,
    long long sA, long long sB, long long sC)
{
  int bz = blockIdx.z;
  Af += (long long)bz * sA;
  Bf += (long long)bz * sB;
  C  += (long long)bz * sC;
  int mt = blockIdx.y;
  int nt0 = blockIdx.x * 4;
  int lane = threadIdx.x & 31;

  v8f acc0 = {}, acc1 = {}, acc2 = {}, acc3 = {};
  const half_t* Ap = Af + ((long long)mt * Ksteps) * 512 + lane * 16;
#pragma unroll 2
  for (int ks = 0; ks < Ksteps; ++ks) {
    const half_t* Bp = Bf + ((long long)ks * Ntiles + nt0) * 512 + lane * 16;
    if (ks + 1 < Ksteps) {
      __builtin_prefetch(Bp + (long long)Ntiles * 512, 0, 3);
      __builtin_prefetch(Ap + (ks + 1) * 512, 0, 3);
    }
    v16h a  = *(const v16h*)(Ap + ks * 512);
    v16h b0 = *(const v16h*)(Bp);
    v16h b1 = *(const v16h*)(Bp + 512);
    v16h b2 = *(const v16h*)(Bp + 1024);
    v16h b3 = *(const v16h*)(Bp + 1536);
    acc0 = WMMA_F16(a, b0, acc0);
    acc1 = WMMA_F16(a, b1, acc1);
    acc2 = WMMA_F16(a, b2, acc2);
    acc3 = WMMA_F16(a, b3, acc3);
  }
  int nl = lane & 15;
  long long mrow = (long long)(mt * 16 + 8 * (lane >> 4));
  v8f accs[4] = {acc0, acc1, acc2, acc3};
#pragma unroll
  for (int i = 0; i < 4; ++i) {
    int n = (nt0 + i) * 16 + nl;
    float bv = bias ? bias[n] : 0.0f;
    float* Cp = C + mrow * ldc + n;
#pragma unroll
    for (int v = 0; v < 8; ++v) Cp[(long long)v * ldc] = accs[i][v] + bv;
  }
}

// ---------------- zero fill (uint4 granularity) ----------------
__global__ __launch_bounds__(256) void zero_u4(uint4* __restrict__ p, int n) {
  int i = blockIdx.x * 256 + threadIdx.x;
  if (i < n) p[i] = make_uint4(0u, 0u, 0u, 0u);
}

// ---------------- weight packing: f32 row-major -> B-fragment layout f16 -------------
__global__ __launch_bounds__(256) void pack_bfrag(const float* __restrict__ src,
                                                  half_t* __restrict__ dst,
                                                  int rows, int cols, int colofs, int Ntiles)
{
  int i = blockIdx.x * 256 + threadIdx.x;
  if (i >= rows * cols) return;
  int k = i / cols, c = i - k * cols;
  int n = colofs + c;
  dst[(((long long)(k >> 5)) * Ntiles + (n >> 4)) * 512 + b_frag_idx(k & 31, n & 15)] =
      (half_t)src[i];
}

// ---------------- LN(s) -> A-fragment layout f16 ([72 mt][8 ks][512]) ----------------
__global__ __launch_bounds__(256) void ln_s_kernel(const float* __restrict__ s,
                                                   const float* __restrict__ scale,
                                                   const float* __restrict__ bias,
                                                   half_t* __restrict__ sA)
{
  __shared__ float red[32];
  int l = blockIdx.x, t = threadIdx.x;
  float x = s[(long long)l * CS_ + t];
  float mu = block_reduce_sum(x, red, 8) * (1.0f / CS_);
  float d = x - mu;
  float var = block_reduce_sum(d * d, red, 8) * (1.0f / CS_);
  float y = d * rsqrtf(var + 1e-5f) * scale[t] + bias[t];
  sA[(((long long)(l >> 4)) * 8 + (t >> 5)) * 512 + a_frag_idx(l & 15, t & 31)] = (half_t)y;
}

// ---- LN(z) -> per-residue B-fragments ([384 i][12 ks][8 nt][512]) and b_res = z_ln@w_b
__global__ __launch_bounds__(128) void ln_z_kernel(const float* __restrict__ z,
                                                   const float* __restrict__ scale,
                                                   const float* __restrict__ bias,
                                                   const float* __restrict__ w_b,
                                                   half_t* __restrict__ zB,
                                                   float* __restrict__ b_res)
{
  __shared__ float red[32];
  long long row = blockIdx.x;     // i*NRES + j
  int t = threadIdx.x;
  float x = z[row * CZ_ + t];
  float mu = block_reduce_sum(x, red, 4) * (1.0f / CZ_);
  float d = x - mu;
  float var = block_reduce_sum(d * d, red, 4) * (1.0f / CZ_);
  float y = d * rsqrtf(var + 1e-5f) * scale[t] + bias[t];
  int i = (int)(row / NRES_), j = (int)(row - (long long)i * NRES_);
  zB[((((long long)i * 12 + (j >> 5)) * 8) + (t >> 4)) * 512 + b_frag_idx(j & 31, t & 15)] =
      (half_t)y;
  // batched 8-head reduction: one shuffle tree, one barrier
  float part[NH];
#pragma unroll
  for (int h = 0; h < NH; ++h) part[h] = y * w_b[t * NH + h];
  for (int o = 16; o > 0; o >>= 1) {
#pragma unroll
    for (int h = 0; h < NH; ++h) part[h] += __shfl_down(part[h], o, 32);
  }
  if ((t & 31) == 0) {
#pragma unroll
    for (int h = 0; h < NH; ++h) red[(t >> 5) * NH + h] = part[h];
  }
  __syncthreads();
  if (t < NH) b_res[row * NH + t] = red[t] + red[NH + t] + red[2 * NH + t] + red[3 * NH + t];
}

// ---------------- build pre-swizzled QK/V fragments + rank-1 biases ------------------
__global__ __launch_bounds__(256) void build_feats(
    const float* __restrict__ proj, const float* __restrict__ rot,
    const float* __restrict__ trans, const float* __restrict__ mask,
    const float* __restrict__ head_weights,
    half_t* __restrict__ Afrag, half_t* __restrict__ Bfrag, half_t* __restrict__ Vfrag,
    float* __restrict__ bias_row, float* __restrict__ bias_col)
{
  int idx = blockIdx.x * 256 + threadIdx.x;
  if (idx >= L_TOT * NH) return;
  int l = idx / NH, h = idx - l * NH;
  const float* pr = proj + (long long)l * NPROJ_LD;
  float R[9], t3[3];
#pragma unroll
  for (int i = 0; i < 9; ++i) R[i] = rot[l * 9 + i];
#pragma unroll
  for (int i = 0; i < 3; ++i) t3[i] = trans[l * 3 + i];
  float hw = log1pf(expf(head_weights[h])) * sqrtf(1.0f / 108.0f);
  const float scale_qk = 0.14433756729740643f;     // sqrt(1/48)
  long long abase = (((long long)h * MT_L + (l >> 4)) * 2) * 512;
  int m = l & 15;
#pragma unroll
  for (int c = 0; c < CH_; ++c) {
    Afrag[abase + a_frag_idx(m, c)] = (half_t)(pr[h * CH_ + c] * scale_qk);
    Bfrag[abase + b_frag_idx(c, m)] = (half_t)(pr[128 + h * CH_ + c]);
  }
  float qq = 0.0f, kk = 0.0f;
#pragma unroll
  for (int p = 0; p < PQ_; ++p) {
    float qp[3], kp[3];
#pragma unroll
    for (int d = 0; d < 3; ++d) { qp[d] = pr[384 + h * 24 + p * 3 + d]; kp[d] = pr[576 + h * 24 + p * 3 + d]; }
#pragma unroll
    for (int i = 0; i < 3; ++i) {
      float qg = R[i*3+0]*qp[0] + R[i*3+1]*qp[1] + R[i*3+2]*qp[2] + t3[i];
      float kg = R[i*3+0]*kp[0] + R[i*3+1]*kp[1] + R[i*3+2]*kp[2] + t3[i];
      qq += qg * qg; kk += kg * kg;
      int kf = 16 + p * 3 + i;              // 16..39
      int ks = kf >> 5, kl = kf & 31;
      Afrag[abase + ks * 512 + a_frag_idx(m, kl)] = (half_t)(qg * hw);
      Bfrag[abase + ks * 512 + b_frag_idx(kl, m)] = (half_t)kg;
    }
  }
#pragma unroll
  for (int kf = 40; kf < 64; ++kf) {        // zero padding
    int ks = kf >> 5, kl = kf & 31;
    Afrag[abase + ks * 512 + a_frag_idx(m, kl)] = (half_t)0.0f;
    Bfrag[abase + ks * 512 + b_frag_idx(kl, m)] = (half_t)0.0f;
  }
  // V features: [v(16) | v_g(36) | pad(12)] as B fragments (k = l%32, n = feature%16)
  float vg[PV_][3];
#pragma unroll
  for (int p = 0; p < PV_; ++p) {
    float vp[3];
#pragma unroll
    for (int d = 0; d < 3; ++d) vp[d] = pr[768 + h * 36 + p * 3 + d];
#pragma unroll
    for (int i = 0; i < 3; ++i)
      vg[p][i] = R[i*3+0]*vp[0] + R[i*3+1]*vp[1] + R[i*3+2]*vp[2] + t3[i];
  }
  long long vbase = (((long long)h * JK_L + (l >> 5)) * 4) * 512;
  int klv = l & 31;
#pragma unroll
  for (int f = 0; f < 64; ++f) {
    float val = 0.0f;
    if (f < 16) val = pr[256 + h * CH_ + f];
    else if (f < 52) val = vg[(f - 16) / 3][(f - 16) % 3];
    Vfrag[vbase + (f >> 4) * 512 + b_frag_idx(klv, f & 15)] = (half_t)val;
  }
  bias_row[h * L_TOT + l] = -0.5f * hw * qq;
  bias_col[h * L_TOT + l] = -0.5f * hw * kk + (mask[l] - 1.0f) * 1e9f;
}

// ---------------- fused attention: logits (WMMA) -> softmax in LDS -> attn@V (WMMA) --
#define K4_LOGITS_BYTES (32 * L_TOT * 4)
#define K4_AFR_BYTES    (2 * JK_L * 512 * 2)
#define K4_BRES_BYTES   (12 * NRES_ * 4)
#define K4_SMEM         (K4_LOGITS_BYTES + K4_AFR_BYTES + K4_BRES_BYTES + 2048)

__global__ __launch_bounds__(256, 1) void attention_kernel(
    const half_t* __restrict__ Afrag, const half_t* __restrict__ Bfrag,
    const half_t* __restrict__ Vfrag,
    const float* __restrict__ bias_row, const float* __restrict__ bias_col,
    const float* __restrict__ b_res,
    half_t* __restrict__ attnresA, float* __restrict__ out_av)
{
  extern __shared__ char smem[];
  float*  logits = (float*)smem;                                   // 32 x 1152
  half_t* afr    = (half_t*)(smem + K4_LOGITS_BYTES);              // 2 x 36 x 512
  float*  bres_s = (float*)(smem + K4_LOGITS_BYTES + K4_AFR_BYTES);// <=12 x 384
  float*  red    = (float*)(smem + K4_LOGITS_BYTES + K4_AFR_BYTES + K4_BRES_BYTES);
  float*  rowmax = red + 32 * 8;
  float*  rowsum = rowmax + 32;

  int lt = blockIdx.x, h = blockIdx.y;
  int wave = threadIdx.x >> 5, lane = threadIdx.x & 31;
  int mt = wave >> 2;
  int nloc = lane & 15, mbase = (lane >> 4) * 8;

  // stage this block's b_res slice (rows resbase..resbase+nresblk-1, head h) into LDS
  int resbase = (lt * 32) / 3;
  int nresblk = ((lt * 32 + 31) / 3) - resbase + 1;   // <= 12
  for (int i = threadIdx.x; i < nresblk * NRES_; i += 256) {
    int rr = i / NRES_, c = i - rr * NRES_;
    bres_s[i] = b_res[((long long)(resbase + rr) * NRES_ + c) * NH + h];
  }

  long long ab = (((long long)h * MT_L + (lt * 2 + mt)) * 2) * 512 + lane * 16;
  v16h a0 = *(const v16h*)(Afrag + ab);
  v16h a1 = *(const v16h*)(Afrag + ab + 512);

  float brow[8]; int rl[8];
#pragma unroll
  for (int v = 0; v < 8; ++v) {
    int lg = lt * 32 + mt * 16 + mbase + v;
    brow[v] = bias_row[h * L_TOT + lg];
    rl[v] = (lg / 3 - resbase) * NRES_;
  }
  __syncthreads();

  // phase 1: logits
  for (int jt = (wave & 3); jt < MT_L; jt += 4) {
    const half_t* Bb = Bfrag + (((long long)h * MT_L + jt) * 2) * 512 + lane * 16;
    if (jt + 4 < MT_L) __builtin_prefetch(Bb + 4 * 1024, 0, 3);
    v16h b0 = *(const v16h*)(Bb);
    v16h b1 = *(const v16h*)(Bb + 512);
    v8f acc = {};
    acc = WMMA_F16(a0, b0, acc);
    acc = WMMA_F16(a1, b1, acc);
    int j = jt * 16 + nloc;
    float bcol = bias_col[h * L_TOT + j];
    int rj = j / 3;
#pragma unroll
    for (int v = 0; v < 8; ++v) {
      float x = acc[v] + brow[v] + bcol + 0.57735026918962576f * bres_s[rl[v] + rj];
      logits[(mt * 16 + mbase + v) * L_TOT + j] = x;
    }
  }
  __syncthreads();

  // phase 2: softmax (8 threads per row)
  int row = threadIdx.x >> 3, sub = threadIdx.x & 7;
  float mx = -1e30f;
  for (int j = sub; j < L_TOT; j += 8) mx = fmaxf(mx, logits[row * L_TOT + j]);
  red[row * 8 + sub] = mx;
  __syncthreads();
  if (sub == 0) {
    float m2 = red[row * 8];
#pragma unroll
    for (int u = 1; u < 8; ++u) m2 = fmaxf(m2, red[row * 8 + u]);
    rowmax[row] = m2;
  }
  __syncthreads();
  float rm = rowmax[row];
  float sm = 0.0f;
  for (int j = sub; j < L_TOT; j += 8) {
    float e = __expf(logits[row * L_TOT + j] - rm);
    logits[row * L_TOT + j] = e;
    sm += e;
  }
  red[row * 8 + sub] = sm;
  __syncthreads();
  if (sub == 0) {
    float s2 = 0.0f;
#pragma unroll
    for (int u = 0; u < 8; ++u) s2 += red[row * 8 + u];
    rowsum[row] = s2;
  }
  __syncthreads();
  float inv = 1.0f / rowsum[row];
  // pack normalized attn directly into A-fragment layout in LDS
  for (int j = sub; j < L_TOT; j += 8) {
    float a = logits[row * L_TOT + j] * inv;
    int mt2 = row >> 4, ml = row & 15, jk = j >> 5, kl = j & 31;
    afr[((mt2 * JK_L + jk) << 9) + a_frag_idx(ml, kl)] = (half_t)a;
  }
  __syncthreads();

  // attn_res (sum over RPR=3) -> per-residue A-fragment layout (rows = ll*8+h, K = n)
  for (int i = threadIdx.x; i < 32 * NRES_; i += 256) {
    int rw = i / NRES_, n = i - rw * NRES_;
    float s3 = logits[rw * L_TOT + n * 3] + logits[rw * L_TOT + n * 3 + 1]
             + logits[rw * L_TOT + n * 3 + 2];
    float a = s3 / rowsum[rw];
    int lg = lt * 32 + rw, r = lg / 3, ll = lg - r * 3;
    int mrow = ll * NH + h;      // 0..23
    attnresA[((((long long)r * 2 + (mrow >> 4)) * 12) + (n >> 5)) * 512 +
             a_frag_idx(mrow & 15, n & 31)] = (half_t)a;
  }

  // phase 3: attn @ [v | v_g] (K = 1152, N = 64)
  int ft = wave & 3;
  v8f accv = {};
  for (int jk = 0; jk < JK_L; ++jk) {
    const half_t* Vb = Vfrag + (((long long)h * JK_L + jk) * 4 + ft) * 512 + lane * 16;
    if (jk + 1 < JK_L) __builtin_prefetch(Vb + 4 * 512, 0, 3);
    v16h af = *(const v16h*)(afr + ((mt * JK_L + jk) << 9) + lane * 16);
    v16h bf = *(const v16h*)(Vb);
    accv = WMMA_F16(af, bf, accv);
  }
  int f = ft * 16 + nloc;
#pragma unroll
  for (int v = 0; v < 8; ++v) {
    int lg = lt * 32 + mt * 16 + mbase + v;
    out_av[((long long)lg * NH + h) * 64 + f] = accv[v];
  }
}

// ------ concat builder -> A-fragment layout ([72 mt][48 ks][512]) --------------------
__global__ __launch_bounds__(256) void build_cat(
    const float* __restrict__ out_av, const float* __restrict__ o_pair,
    const float* __restrict__ rot, const float* __restrict__ trans,
    half_t* __restrict__ catA)
{
  int idx = blockIdx.x * 256 + threadIdx.x;
  if (idx >= L_TOT * NH) return;
  int l = idx / NH, h = idx - l * NH;
  const float* av = out_av + ((long long)l * NH + h) * 64;
  float R[9], t3[3];
#pragma unroll
  for (int i = 0; i < 9; ++i) R[i] = rot[l * 9 + i];
#pragma unroll
  for (int i = 0; i < 3; ++i) t3[i] = trans[l * 3 + i];
  long long cb = ((long long)(l >> 4)) * 48 * 512;
  int ml = l & 15;
  int fb = h * 192;   // feature base within K = 1536
  auto put = [&](int f, float val) {
    int kf = fb + f;
    catA[cb + (kf >> 5) * 512 + a_frag_idx(ml, kf & 31)] = (half_t)val;
  };
#pragma unroll
  for (int c = 0; c < CH_; ++c) put(c, av[c]);
#pragma unroll
  for (int p = 0; p < PV_; ++p) {
    float g0 = av[16 + p*3 + 0] - t3[0];
    float g1 = av[16 + p*3 + 1] - t3[1];
    float g2 = av[16 + p*3 + 2] - t3[2];
    float o0 = R[0]*g0 + R[3]*g1 + R[6]*g2;   // rot^T
    float o1 = R[1]*g0 + R[4]*g1 + R[7]*g2;
    float o2 = R[2]*g0 + R[5]*g1 + R[8]*g2;
    put(16 + p*3 + 0, o0);
    put(16 + p*3 + 1, o1);
    put(16 + p*3 + 2, o2);
    put(52 + p, sqrtf(o0*o0 + o1*o1 + o2*o2 + 1e-8f));
  }
  int r = l / 3, ll = l - r * 3;
  const float* op = o_pair + ((long long)r * 32 + ll * NH + h) * CZ_;  // padded 32 rows
#pragma unroll
  for (int c = 0; c < CZ_; ++c) put(64 + c, op[c]);
}

// ---------------- host-side launch ----------------
extern "C" void kernel_launch(void* const* d_in, const int* in_sizes, int n_in,
                              void* d_out, int out_size, void* d_ws, size_t ws_size,
                              hipStream_t stream)
{
  const float* s        = (const float*)d_in[0];
  const float* z        = (const float*)d_in[1];
  const float* rot      = (const float*)d_in[2];
  const float* trans    = (const float*)d_in[3];
  const float* mask     = (const float*)d_in[4];
  const float* ln_s_sc  = (const float*)d_in[5];
  const float* ln_s_b   = (const float*)d_in[6];
  const float* ln_z_sc  = (const float*)d_in[7];
  const float* ln_z_b   = (const float*)d_in[8];
  const float* w_q      = (const float*)d_in[9];
  const float* w_k      = (const float*)d_in[10];
  const float* w_v      = (const float*)d_in[11];
  const float* w_qp     = (const float*)d_in[12];
  const float* w_kp     = (const float*)d_in[13];
  const float* w_vp     = (const float*)d_in[14];
  const float* w_b      = (const float*)d_in[15];
  const float* head_w   = (const float*)d_in[16];
  const float* w_out    = (const float*)d_in[17];
  const float* b_out    = (const float*)d_in[18];
  float* out = (float*)d_out;

  char* p = (char*)d_ws;
  auto alloc = [&](size_t bytes) -> void* {
    void* r = (void*)p;
    p += (bytes + 1023) & ~(size_t)1023;
    return r;
  };
  // B-fragments: WallB [8 ks][68 nt][512] (padded), WoutB [48 ks][16 nt][512]
  half_t* WallB    = (half_t*)alloc((size_t)8 * 68 * 512 * 2);
  half_t* WoutB    = (half_t*)alloc((size_t)48 * 16 * 512 * 2);
  half_t* sA       = (half_t*)alloc((size_t)MT_L * 8 * 512 * 2);        // A-frags of s_ln
  float*  proj     = (float*) alloc((size_t)L_TOT * NPROJ_LD * 4);
  half_t* Afrag    = (half_t*)alloc((size_t)NH * MT_L * 2 * 512 * 2);
  half_t* Bfrag    = (half_t*)alloc((size_t)NH * MT_L * 2 * 512 * 2);
  half_t* Vfrag    = (half_t*)alloc((size_t)NH * JK_L * 4 * 512 * 2);
  float*  biasR    = (float*) alloc((size_t)NH * L_TOT * 4);
  float*  biasC    = (float*) alloc((size_t)NH * L_TOT * 4);
  half_t* zB       = (half_t*)alloc((size_t)NRES_ * 12 * 8 * 512 * 2);  // per-res B-frags
  float*  b_res    = (float*) alloc((size_t)NRES_ * NRES_ * NH * 4);
  half_t* attnresA = (half_t*)alloc((size_t)NRES_ * 2 * 12 * 512 * 2);  // per-res A-frags
  float*  out_av   = (float*) alloc((size_t)L_TOT * NH * 64 * 4);
  float*  o_pair   = (float*) alloc((size_t)NRES_ * 32 * CZ_ * 4);      // padded 32 rows
  half_t* catA     = (half_t*)alloc((size_t)MT_L * 48 * 512 * 2);

  auto g1 = [](int n) { return dim3((n + 255) / 256); };

  // 0) zero padded weight fragments, then pack weights into B-fragment layouts
  {
    int n16 = (int)((size_t)8 * 68 * 512 / 8);
    zero_u4<<<g1(n16), 256, 0, stream>>>((uint4*)WallB, n16);
  }
  pack_bfrag<<<g1(CS_*128), 256, 0, stream>>>(w_q,  WallB, CS_, 128, 0,   68);
  pack_bfrag<<<g1(CS_*128), 256, 0, stream>>>(w_k,  WallB, CS_, 128, 128, 68);
  pack_bfrag<<<g1(CS_*128), 256, 0, stream>>>(w_v,  WallB, CS_, 128, 256, 68);
  pack_bfrag<<<g1(CS_*192), 256, 0, stream>>>(w_qp, WallB, CS_, 192, 384, 68);
  pack_bfrag<<<g1(CS_*192), 256, 0, stream>>>(w_kp, WallB, CS_, 192, 576, 68);
  pack_bfrag<<<g1(CS_*288), 256, 0, stream>>>(w_vp, WallB, CS_, 288, 768, 68);
  pack_bfrag<<<g1(CATK*CS_), 256, 0, stream>>>(w_out, WoutB, CATK, CS_, 0, 16);

  // 1) LN(s) -> A-fragments
  ln_s_kernel<<<L_TOT, 256, 0, stream>>>(s, ln_s_sc, ln_s_b, sA);

  // 2) projections: proj = s_ln @ Wall (1152 x 256 x 1088pad), guard-free fragment GEMM
  gemm_frag<<<dim3(17, MT_L, 1), 32, 0, stream>>>(
      sA, WallB, proj, nullptr, 68, 8, NPROJ_LD, 0, 0, 0);

  // 3) feature builder (pre-swizzled WMMA fragments + rank-1 biases)
  build_feats<<<g1(L_TOT * NH), 256, 0, stream>>>(
      proj, rot, trans, mask, head_w, Afrag, Bfrag, Vfrag, biasR, biasC);

  // 4) zero-fill padded attn_res fragments (rows 24..31 stay zero)
  {
    int n16 = (int)((size_t)NRES_ * 2 * 12 * 512 / 8);
    zero_u4<<<g1(n16), 256, 0, stream>>>((uint4*)attnresA, n16);
  }

  // 5) LN(z) -> per-residue B-fragments + b_res
  ln_z_kernel<<<NRES_ * NRES_, 128, 0, stream>>>(z, ln_z_sc, ln_z_b, w_b, zB, b_res);

  // 6) fused attention
  attention_kernel<<<dim3(JK_L, NH), 256, K4_SMEM, stream>>>(
      Afrag, Bfrag, Vfrag, biasR, biasC, b_res, attnresA, out_av);

  // 7) o_pair: per residue, (32pad x 384) @ (384 x 128) batched fragment GEMM
  gemm_frag<<<dim3(2, 2, NRES_), 32, 0, stream>>>(
      attnresA, zB, o_pair, nullptr, 8, 12, CZ_,
      (long long)2 * 12 * 512, (long long)12 * 8 * 512, (long long)32 * CZ_);

  // 8) concat epilogue -> A-fragments
  build_cat<<<g1(L_TOT * NH), 256, 0, stream>>>(out_av, o_pair, rot, trans, catA);

  // 9) output projection: out = cat @ w_out + b_out (1152 x 1536 x 256)
  gemm_frag<<<dim3(4, MT_L, 1), 32, 0, stream>>>(
      catA, WoutB, out, b_out, 16, 48, CS_, 0, 0, 0);

  (void)in_sizes; (void)n_in; (void)out_size; (void)ws_size;
}
